// RAFT_27127013442005
// MI455X (gfx1250) — matrix operand order repaired and checked by reading, not compiled, not run on hardware
//
#include <hip/hip_runtime.h>
#include <cstddef>
#include <cstdint>

// ---------------------------------------------------------------------------
// Types for CDNA5 WMMA (wave32)
// ---------------------------------------------------------------------------
typedef _Float16 h8    __attribute__((ext_vector_type(8)));
typedef _Float16 v16h  __attribute__((ext_vector_type(16)));
typedef float    v8f   __attribute__((ext_vector_type(8)));

#define TSZ 64      // 64x64 macro tile per 256-thread block (8 wave32 waves)
#define KS  32      // K step per WMMA
#define LDK 40      // padded f16 LDS row stride (halfs) -> 80B rows, 16B aligned

// ---------------------------------------------------------------------------
// k -> (cin, kh, kw) decode LUT: one entry per reduction index, packed as
// c | kh<<16 | kw<<24. Removes the div/mod chains from the conv inner loop.
// ---------------------------------------------------------------------------
__global__ void kdec_k(int* __restrict__ kd, int K, int Cin, int KW)
{
  int k = blockIdx.x * 256 + threadIdx.x;
  if (k >= K) return;
  int c  = k % Cin;
  int r  = k / Cin;
  int kw = r % KW;
  int kh = r / KW;
  kd[k] = c | (kh << 16) | (kw << 24);
}

// ---------------------------------------------------------------------------
// Implicit-GEMM convolution (NHWC in, HWIO weights), f16 WMMA, f32 accum.
// M = B*Ho*Wo, N = Cout, K = KH*KW*Cin. M must be a multiple of 64 (true for
// every conv in this net); N and K are bounds-checked.
// Register-pipelined: next K-tile's global loads are issued before the current
// tile's WMMA, hiding global latency behind the matrix ops.
// ---------------------------------------------------------------------------
__global__ __launch_bounds__(256) void conv_gemm_k(
    const float* __restrict__ in, const float* __restrict__ wgt,
    const float* __restrict__ bias, const int* __restrict__ kdec,
    float* __restrict__ out,
    int B, int H, int W, int Cin, int Ho, int Wo, int Cout,
    int stride, int padh, int padw, int K, int relu)
{
  __shared__ _Float16 As[TSZ * LDK];
  __shared__ _Float16 Bs[TSZ * LDK];

  const int M  = B * Ho * Wo;
  const int tid = threadIdx.x;
  const int m0 = blockIdx.x * TSZ;
  const int n0 = blockIdx.y * TSZ;

  // loader coordinates: each thread fills 8 contiguous k of one tile row
  const int lrow = tid >> 2;
  const int lk   = (tid & 3) * 8;

  // decode the output pixel for the A row this thread loads (m0+lrow < M)
  const int am   = m0 + lrow;
  const int ab   = am / (Ho * Wo);
  const int arem = am % (Ho * Wo);
  const int aho  = arem / Wo;
  const int awo  = arem % Wo;
  const int hbase = aho * stride - padh;
  const int wbase = awo * stride - padw;
  const float* inb = in + (size_t)ab * H * W * Cin;
  const int nB = n0 + lrow;

  // wave/tile mapping: 8 waves -> 4x4 grid of 16x16 tiles, 2 tiles per wave
  const int wid  = tid >> 5;
  const int lane = tid & 31;
  const int hid  = lane >> 4;   // lane half (K-half select per ISA layout)
  const int l16  = lane & 15;
  const int tm   = wid >> 1;
  const int tn0  = (wid & 1) * 2;

  float ar8[8], br8[8];

  auto load_tiles = [&](int k0) {
#pragma unroll
    for (int j = 0; j < 8; ++j) {
      int k = k0 + lk + j;
      float v = 0.f;
      if (k < K) {
        int d   = kdec[k];
        int c   = d & 0xFFFF;
        int hi  = hbase + ((d >> 16) & 0xFF);
        int wi  = wbase + ((d >> 24) & 0xFF);
        if (hi >= 0 && hi < H && wi >= 0 && wi < W)
          v = inb[((size_t)hi * W + wi) * Cin + c];
      }
      ar8[j] = v;
      br8[j] = (k < K && nB < Cout) ? wgt[(size_t)k * Cout + nB] : 0.f;
    }
  };

  v8f acc0 = {}; v8f acc1 = {};
  load_tiles(0);

  for (int k0 = 0; k0 < K; k0 += KS) {
    // ---- commit staged registers to LDS (f32 -> f16) ----
#pragma unroll
    for (int j = 0; j < 8; ++j) {
      As[lrow * LDK + lk + j] = (_Float16)ar8[j];
      Bs[lrow * LDK + lk + j] = (_Float16)br8[j];
    }
    __syncthreads();

    // ---- issue next tile's global loads (overlap with WMMA below) ----
    if (k0 + KS < K) load_tiles(k0 + KS);

    // ---- fragments per the 16-bit WMMA VGPR layouts ----
    const _Float16* ar = &As[(tm * 16 + l16) * LDK];
    h8 a0 = *(const h8*)&ar[hid * 8];
    h8 a1 = *(const h8*)&ar[16 + hid * 8];
    v16h af;
#pragma unroll
    for (int i = 0; i < 8; ++i) { af[i] = a0[i]; af[i + 8] = a1[i]; }

    const _Float16* br0 = &Bs[(tn0 * 16 + l16) * LDK + hid * 8];
    h8 b00 = *(const h8*)&br0[0];
    h8 b01 = *(const h8*)&br0[16];
    v16h bf0;
#pragma unroll
    for (int i = 0; i < 8; ++i) { bf0[i] = b00[i]; bf0[i + 8] = b01[i]; }

    const _Float16* br1 = &Bs[((tn0 + 1) * 16 + l16) * LDK + hid * 8];
    h8 b10 = *(const h8*)&br1[0];
    h8 b11 = *(const h8*)&br1[16];
    v16h bf1;
#pragma unroll
    for (int i = 0; i < 8; ++i) { bf1[i] = b10[i]; bf1[i + 8] = b11[i]; }

    acc0 = __builtin_amdgcn_wmma_f32_16x16x32_f16(false, af, false, bf0,
                                                  (short)0, acc0, false, false);
    acc1 = __builtin_amdgcn_wmma_f32_16x16x32_f16(false, af, false, bf1,
                                                  (short)0, acc1, false, false);
    __syncthreads();
  }

  // ---- epilogue: bias (+ optional ReLU), scatter to NHWC ----
#pragma unroll
  for (int r = 0; r < 8; ++r) {
    int m = m0 + tm * 16 + hid * 8 + r;
    if (m >= M) continue;
    int b  = m / (Ho * Wo);
    int rm = m % (Ho * Wo);
    int ho = rm / Wo;
    int wo = rm % Wo;
    size_t obase = (((size_t)b * Ho + ho) * Wo + wo) * Cout;
    int n = n0 + tn0 * 16 + l16;
    if (n < Cout) {
      float v = acc0[r] + bias[n];
      if (relu) v = fmaxf(v, 0.f);
      out[obase + n] = v;
    }
    int n1 = n + 16;
    if (n1 < Cout) {
      float v = acc1[r] + bias[n1];
      if (relu) v = fmaxf(v, 0.f);
      out[obase + n1] = v;
    }
  }
}

// ---------------------------------------------------------------------------
// Correlation volume: per batch, out[m][n] = dot(f1[m,:], f2[n,:]) * scale.
// A rows and B rows are both contiguous in K (M = N = P = 3072, K = 256, all
// multiples of the tile sizes: no bounds handling). float4 (b128) global
// loads, register-pipelined staging, f16 WMMA with f32 accumulate.
// ---------------------------------------------------------------------------
__global__ __launch_bounds__(256) void corr_gemm_k(
    const float* __restrict__ f1, const float* __restrict__ f2,
    float* __restrict__ out, int P, int K, float scale)
{
  __shared__ _Float16 As[TSZ * LDK];
  __shared__ _Float16 Bs[TSZ * LDK];

  const int tid = threadIdx.x;
  const int m0 = blockIdx.x * TSZ;
  const int n0 = blockIdx.y * TSZ;
  const int b  = blockIdx.z;
  const float* A  = f1 + (size_t)b * P * K;
  const float* Bm = f2 + (size_t)b * P * K;

  const int lrow = tid >> 2;
  const int lk   = (tid & 3) * 8;
  const int wid  = tid >> 5;
  const int lane = tid & 31;
  const int hid  = lane >> 4;
  const int l16  = lane & 15;
  const int tm   = wid >> 1;
  const int tn0  = (wid & 1) * 2;

  const float* arow = &A[(size_t)(m0 + lrow) * K + lk];
  const float* brow = &Bm[(size_t)(n0 + lrow) * K + lk];

  float4 a0v, a1v, b0v, b1v;
  auto load_tiles = [&](int k0) {
    a0v = *(const float4*)&arow[k0];
    a1v = *(const float4*)&arow[k0 + 4];
    b0v = *(const float4*)&brow[k0];
    b1v = *(const float4*)&brow[k0 + 4];
  };

  v8f acc0 = {}; v8f acc1 = {};
  load_tiles(0);

  for (int k0 = 0; k0 < K; k0 += KS) {
    _Float16* as = &As[lrow * LDK + lk];
    as[0] = (_Float16)a0v.x; as[1] = (_Float16)a0v.y;
    as[2] = (_Float16)a0v.z; as[3] = (_Float16)a0v.w;
    as[4] = (_Float16)a1v.x; as[5] = (_Float16)a1v.y;
    as[6] = (_Float16)a1v.z; as[7] = (_Float16)a1v.w;
    _Float16* bs = &Bs[lrow * LDK + lk];
    bs[0] = (_Float16)b0v.x; bs[1] = (_Float16)b0v.y;
    bs[2] = (_Float16)b0v.z; bs[3] = (_Float16)b0v.w;
    bs[4] = (_Float16)b1v.x; bs[5] = (_Float16)b1v.y;
    bs[6] = (_Float16)b1v.z; bs[7] = (_Float16)b1v.w;
    __syncthreads();

    if (k0 + KS < K) load_tiles(k0 + KS);

    const _Float16* ar = &As[(tm * 16 + l16) * LDK];
    h8 a0 = *(const h8*)&ar[hid * 8];
    h8 a1 = *(const h8*)&ar[16 + hid * 8];
    v16h af;
#pragma unroll
    for (int i = 0; i < 8; ++i) { af[i] = a0[i]; af[i + 8] = a1[i]; }

    const _Float16* br0 = &Bs[(tn0 * 16 + l16) * LDK + hid * 8];
    h8 b00 = *(const h8*)&br0[0];
    h8 b01 = *(const h8*)&br0[16];
    v16h bf0;
#pragma unroll
    for (int i = 0; i < 8; ++i) { bf0[i] = b00[i]; bf0[i + 8] = b01[i]; }

    const _Float16* br1 = &Bs[((tn0 + 1) * 16 + l16) * LDK + hid * 8];
    h8 b10 = *(const h8*)&br1[0];
    h8 b11 = *(const h8*)&br1[16];
    v16h bf1;
#pragma unroll
    for (int i = 0; i < 8; ++i) { bf1[i] = b10[i]; bf1[i + 8] = b11[i]; }

    acc0 = __builtin_amdgcn_wmma_f32_16x16x32_f16(false, af, false, bf0,
                                                  (short)0, acc0, false, false);
    acc1 = __builtin_amdgcn_wmma_f32_16x16x32_f16(false, af, false, bf1,
                                                  (short)0, acc1, false, false);
    __syncthreads();
  }

#pragma unroll
  for (int r = 0; r < 8; ++r) {
    int m = m0 + tm * 16 + hid * 8 + r;
    size_t obase = ((size_t)b * P + m) * P;
    int n = n0 + tn0 * 16 + l16;
    out[obase + n]      = acc0[r] * scale;
    out[obase + n + 16] = acc1[r] * scale;
  }
}

// ---------------------------------------------------------------------------
// 2x2 average pool over the trailing (hh, ww) dims of an (N, hh, ww) volume.
// ---------------------------------------------------------------------------
__global__ void pool_k(const float* __restrict__ in, float* __restrict__ out,
                       int N, int hh, int ww)
{
  int oh = hh >> 1, ow = ww >> 1;
  size_t total = (size_t)N * oh * ow;
  size_t t = (size_t)blockIdx.x * blockDim.x + threadIdx.x;
  if (t >= total) return;
  int x = (int)(t % ow);
  int y = (int)((t / ow) % oh);
  int n = (int)(t / ((size_t)ow * oh));
  const float* p = in + (size_t)n * hh * ww;
  float v = 0.25f * (p[(2 * y) * ww + 2 * x]     + p[(2 * y) * ww + 2 * x + 1] +
                     p[(2 * y + 1) * ww + 2 * x] + p[(2 * y + 1) * ww + 2 * x + 1]);
  out[t] = v;
}

// ---------------------------------------------------------------------------
// Instance/batch norm (in-place) + optional ReLU. One block per group:
//   inorm: grid = B*C, npix = H*W        (blockIdx/C = b, blockIdx%C = c)
//   bnorm: grid = C,   npix = B*H*W
// ---------------------------------------------------------------------------
__global__ __launch_bounds__(256) void norm_k(float* __restrict__ x,
                                              const float* __restrict__ g,
                                              const float* __restrict__ bt,
                                              int npix, int C, int relu)
{
  int c = blockIdx.x % C;
  float* base = x + (size_t)(blockIdx.x / C) * npix * C + c;
  __shared__ float s1[256], s2[256];
  float a = 0.f, q = 0.f;
  for (int i = threadIdx.x; i < npix; i += 256) {
    float v = base[(size_t)i * C];
    a += v; q += v * v;
  }
  s1[threadIdx.x] = a; s2[threadIdx.x] = q;
  __syncthreads();
  for (int o = 128; o > 0; o >>= 1) {
    if ((int)threadIdx.x < o) {
      s1[threadIdx.x] += s1[threadIdx.x + o];
      s2[threadIdx.x] += s2[threadIdx.x + o];
    }
    __syncthreads();
  }
  float mean = s1[0] / (float)npix;
  float var  = s2[0] / (float)npix - mean * mean;
  float inv  = rsqrtf(var + 1e-5f) * g[c];
  float sh   = bt[c];
  for (int i = threadIdx.x; i < npix; i += 256) {
    float v = (base[(size_t)i * C] - mean) * inv + sh;
    if (relu) v = fmaxf(v, 0.f);
    base[(size_t)i * C] = v;
  }
}

// ---------------------------------------------------------------------------
// Elementwise helpers
// ---------------------------------------------------------------------------
__global__ void add_relu_k(const float* __restrict__ a, const float* __restrict__ b,
                           float* __restrict__ o, size_t n)
{
  size_t i = (size_t)blockIdx.x * blockDim.x + threadIdx.x;
  if (i < n) o[i] = fmaxf(a[i] + b[i], 0.f);
}

__global__ void concat_k(const float* __restrict__ a, int Ca,
                         const float* __restrict__ b, int Cb,
                         float* __restrict__ o, size_t npix)
{
  int Co = Ca + Cb;
  size_t total = npix * Co;
  size_t t = (size_t)blockIdx.x * blockDim.x + threadIdx.x;
  if (t >= total) return;
  int c = (int)(t % Co);
  size_t p = t / Co;
  o[t] = (c < Ca) ? a[p * Ca + c] : b[p * Cb + (c - Ca)];
}

__global__ void split_cnet_k(const float* __restrict__ cnet,
                             float* __restrict__ net, float* __restrict__ inp,
                             size_t npix)
{
  size_t total = npix * 128;
  size_t t = (size_t)blockIdx.x * blockDim.x + threadIdx.x;
  if (t >= total) return;
  int c = (int)(t % 128);
  size_t p = t / 128;
  net[t] = tanhf(cnet[p * 256 + c]);
  inp[t] = fmaxf(cnet[p * 256 + 128 + c], 0.f);
}

__global__ void rh_k(const float* __restrict__ h, const float* __restrict__ rc,
                     float* __restrict__ rh, size_t n)
{
  size_t i = (size_t)blockIdx.x * blockDim.x + threadIdx.x;
  if (i < n) rh[i] = h[i] / (1.f + expf(-rc[i]));
}

__global__ void gru_out_k(float* __restrict__ h, const float* __restrict__ zc,
                          const float* __restrict__ qc, size_t n)
{
  size_t i = (size_t)blockIdx.x * blockDim.x + threadIdx.x;
  if (i < n) {
    float z = 1.f / (1.f + expf(-zc[i]));
    h[i] = (1.f - z) * h[i] + z * tanhf(qc[i]);
  }
}

__global__ void addflow_k(float* __restrict__ f, const float* __restrict__ d, size_t n)
{
  size_t i = (size_t)blockIdx.x * blockDim.x + threadIdx.x;
  if (i < n) f[i] += d[i];
}

__global__ void fill_k(float* __restrict__ p, float v, size_t n)
{
  size_t i = (size_t)blockIdx.x * blockDim.x + threadIdx.x;
  if (i < n) p[i] = v;
}

// ---------------------------------------------------------------------------
// Correlation-pyramid lookup (bilinear, zero OOB). One thread per (pixel, s).
// ---------------------------------------------------------------------------
__device__ __forceinline__ float bilin(const float* img, int Hl, int Wl,
                                       float x, float y)
{
  float x0f = floorf(x), y0f = floorf(y);
  int x0 = (int)x0f, y0 = (int)y0f;
  float wx1 = x - x0f, wx0 = 1.f - wx1;
  float wy1 = y - y0f, wy0 = 1.f - wy1;
  float acc = 0.f;
#pragma unroll
  for (int dy = 0; dy < 2; ++dy)
#pragma unroll
    for (int dx = 0; dx < 2; ++dx) {
      int xi = x0 + dx, yi = y0 + dy;
      if (xi >= 0 && xi < Wl && yi >= 0 && yi < Hl) {
        float w = (dx ? wx1 : wx0) * (dy ? wy1 : wy0);
        acc += w * img[(size_t)yi * Wl + xi];
      }
    }
  return acc;
}

__global__ void retrieve_k(const float* __restrict__ p0, const float* __restrict__ p1,
                           const float* __restrict__ p2, const float* __restrict__ p3,
                           const float* __restrict__ flow, float* __restrict__ out,
                           int B, int h, int w)
{
  size_t total = (size_t)B * h * w * 81;
  size_t t = (size_t)blockIdx.x * blockDim.x + threadIdx.x;
  if (t >= total) return;
  int s = (int)(t % 81);
  int p = (int)(t / 81);
  int wq = p % w;
  int hq = (p / w) % h;
  float cx = (float)wq + flow[(size_t)p * 2 + 0];
  float cy = (float)hq + flow[(size_t)p * 2 + 1];
  float dx = (float)(s % 9) - 4.f;
  float dy = (float)(s / 9) - 4.f;

  const float* pyr[4] = {p0, p1, p2, p3};
  int Hl = h, Wl = w;
  float scale = 1.f;
  float* ob = out + (size_t)p * 324 + s;
#pragma unroll
  for (int l = 0; l < 4; ++l) {
    const float* img = pyr[l] + (size_t)p * Hl * Wl;
    ob[l * 81] = bilin(img, Hl, Wl, cx * scale + dx, cy * scale + dy);
    Hl >>= 1; Wl >>= 1; scale *= 0.5f;
  }
}

// ---------------------------------------------------------------------------
// Convex upsampling: softmax(0.25*mask) over 9 taps of 8*flow's 3x3 nbhd.
// One thread per (b, h, w, i, j). Output NHWC (B, 8h, 8w, 2).
// ---------------------------------------------------------------------------
__global__ void upsample_k(const float* __restrict__ flow,
                           const float* __restrict__ mask,
                           float* __restrict__ out, int B, int h, int w)
{
  size_t total = (size_t)B * h * w * 64;
  size_t t = (size_t)blockIdx.x * blockDim.x + threadIdx.x;
  if (t >= total) return;
  int ij = (int)(t % 64);
  int p  = (int)(t / 64);
  int j = ij % 8, i = ij / 8;
  int wq = p % w;
  int hq = (p / w) % h;
  int b  = p / (w * h);

  const float* mk = mask + (size_t)p * 576 + ij * 9;
  float e[9];
  float mx = -1e30f;
#pragma unroll
  for (int k = 0; k < 9; ++k) { e[k] = 0.25f * mk[k]; mx = fmaxf(mx, e[k]); }
  float sum = 0.f;
#pragma unroll
  for (int k = 0; k < 9; ++k) { e[k] = expf(e[k] - mx); sum += e[k]; }
  float inv = 1.f / sum;

  float ox = 0.f, oy = 0.f;
#pragma unroll
  for (int ti = 0; ti < 3; ++ti)
#pragma unroll
    for (int tj = 0; tj < 3; ++tj) {
      int hh = hq + ti - 1, ww = wq + tj - 1;
      float fx = 0.f, fy = 0.f;
      if (hh >= 0 && hh < h && ww >= 0 && ww < w) {
        size_t fi = (((size_t)b * h + hh) * w + ww) * 2;
        fx = 8.f * flow[fi];
        fy = 8.f * flow[fi + 1];
      }
      float wgt = e[ti * 3 + tj] * inv;
      ox += wgt * fx;
      oy += wgt * fy;
    }

  int H = h * 8, W = w * 8;
  size_t o = (((size_t)b * H + hq * 8 + i) * W + wq * 8 + j) * 2;
  out[o] = ox;
  out[o + 1] = oy;
}

// ===========================================================================
// Host side
// ===========================================================================
struct BlkP {
  const float *w1, *b1, *g1, *be1, *w2, *b2, *g2, *be2, *wd, *bd, *gd, *bed;
  int has_d, cin, cout, s;
};
struct EncP {
  const float *c1w, *c1b, *n1g, *n1b, *c2w, *c2b;
  BlkP blk[6];
};
struct UpdP {
  const float *cc1w, *cc1b, *cc2w, *cc2b, *cf1w, *cf1b, *cf2w, *cf2b, *cmw, *cmb,
              *fh1w, *fh1b, *fh2w, *fh2b, *m1w, *m1b, *m2w, *m2b,
              *q1w, *q1b, *q2w, *q2b, *r1w, *r1b, *r2w, *r2b,
              *z1w, *z1b, *z2w, *z2b;
};

// jax tree order: dict keys sorted; lists in order. blocks < conv1_b < ... < n1_g
static int parse_enc(void* const* d, int i, EncP& e)
{
  static const int cfg[6][3] = {{64,64,1},{64,64,1},{64,96,2},{96,96,1},{96,128,2},{128,128,1}};
  for (int k = 0; k < 6; ++k) {
    BlkP& b = e.blk[k];
    b.cin = cfg[k][0]; b.cout = cfg[k][1]; b.s = cfg[k][2];
    b.has_d = (b.s != 1 || b.cin != b.cout) ? 1 : 0;
    b.b1 = (const float*)d[i++]; b.b2 = (const float*)d[i++];
    if (b.has_d) b.bd = (const float*)d[i++];
    b.be1 = (const float*)d[i++]; b.be2 = (const float*)d[i++];
    if (b.has_d) b.bed = (const float*)d[i++];
    b.g1 = (const float*)d[i++]; b.g2 = (const float*)d[i++];
    if (b.has_d) b.gd = (const float*)d[i++];
    b.w1 = (const float*)d[i++]; b.w2 = (const float*)d[i++];
    if (b.has_d) b.wd = (const float*)d[i++];
  }
  e.c1b = (const float*)d[i++]; e.c1w = (const float*)d[i++];
  e.c2b = (const float*)d[i++]; e.c2w = (const float*)d[i++];
  e.n1b = (const float*)d[i++]; e.n1g = (const float*)d[i++];
  return i;
}

static int parse_upd(void* const* d, int i, UpdP& u)
{
  u.cc1b=(const float*)d[i++]; u.cc1w=(const float*)d[i++];
  u.cc2b=(const float*)d[i++]; u.cc2w=(const float*)d[i++];
  u.cf1b=(const float*)d[i++]; u.cf1w=(const float*)d[i++];
  u.cf2b=(const float*)d[i++]; u.cf2w=(const float*)d[i++];
  u.cmb =(const float*)d[i++]; u.cmw =(const float*)d[i++];
  u.fh1b=(const float*)d[i++]; u.fh1w=(const float*)d[i++];
  u.fh2b=(const float*)d[i++]; u.fh2w=(const float*)d[i++];
  u.m1b =(const float*)d[i++]; u.m1w =(const float*)d[i++];
  u.m2b =(const float*)d[i++]; u.m2w =(const float*)d[i++];
  u.q1b =(const float*)d[i++]; u.q1w =(const float*)d[i++];
  u.q2b =(const float*)d[i++]; u.q2w =(const float*)d[i++];
  u.r1b =(const float*)d[i++]; u.r1w =(const float*)d[i++];
  u.r2b =(const float*)d[i++]; u.r2w =(const float*)d[i++];
  u.z1b =(const float*)d[i++]; u.z1w =(const float*)d[i++];
  u.z2b =(const float*)d[i++]; u.z2w =(const float*)d[i++];
  return i;
}

static inline dim3 g1d(size_t n) { return dim3((unsigned)((n + 255) / 256)); }

// shared k-decode LUT scratch (stream-ordered reuse across conv calls)
static int* g_kdec = nullptr;

static void run_conv(hipStream_t st, const float* in, const float* w, const float* b,
                     float* out, int B, int H, int W, int Cin, int KH, int KW,
                     int s, int Cout, int relu)
{
  int Ho = (H + s - 1) / s, Wo = (W + s - 1) / s;
  int pth = (Ho - 1) * s + KH - H; if (pth < 0) pth = 0;
  int ptw = (Wo - 1) * s + KW - W; if (ptw < 0) ptw = 0;
  int M = B * Ho * Wo;
  int K = KH * KW * Cin;
  kdec_k<<<dim3((K + 255) / 256), 256, 0, st>>>(g_kdec, K, Cin, KW);
  dim3 g((M + TSZ - 1) / TSZ, (Cout + TSZ - 1) / TSZ);
  conv_gemm_k<<<g, 256, 0, st>>>(in, w, b, g_kdec, out, B, H, W, Cin, Ho, Wo,
                                 Cout, s, pth / 2, ptw / 2, K, relu);
}

static void run_norm(hipStream_t st, float* x, const float* g, const float* bt,
                     int B, int H, int W, int C, int is_bn, int relu)
{
  if (is_bn) norm_k<<<dim3(C), 256, 0, st>>>(x, g, bt, B * H * W, C, relu);
  else       norm_k<<<dim3(B * C), 256, 0, st>>>(x, g, bt, H * W, C, relu);
}

static void run_encoder(hipStream_t st, const float* img, const EncP& e, int is_bn,
                        float* out, int outC, float* bufA, float* bufB, float* bufC,
                        int B)
{
  int H = 384, W = 512;
  run_conv(st, img, e.c1w, e.c1b, bufA, B, H, W, 3, 7, 7, 2, 64, 0);
  H = 192; W = 256; int C = 64;
  run_norm(st, bufA, e.n1g, e.n1b, B, H, W, C, is_bn, 1);

  for (int k = 0; k < 6; ++k) {
    const BlkP& bl = e.blk[k];
    int s = bl.s;
    int Ho = (H + s - 1) / s, Wo = (W + s - 1) / s;
    run_conv(st, bufA, bl.w1, bl.b1, bufB, B, H, W, C, 3, 3, s, bl.cout, 0);
    run_norm(st, bufB, bl.g1, bl.be1, B, Ho, Wo, bl.cout, is_bn, 1);
    run_conv(st, bufB, bl.w2, bl.b2, bufC, B, Ho, Wo, bl.cout, 3, 3, 1, bl.cout, 0);
    run_norm(st, bufC, bl.g2, bl.be2, B, Ho, Wo, bl.cout, is_bn, 1);
    const float* sc = bufA;
    if (bl.has_d) {
      run_conv(st, bufA, bl.wd, bl.bd, bufB, B, H, W, C, 1, 1, s, bl.cout, 0);
      run_norm(st, bufB, bl.gd, bl.bed, B, Ho, Wo, bl.cout, is_bn, 0);
      sc = bufB;
    }
    size_t n = (size_t)B * Ho * Wo * bl.cout;
    add_relu_k<<<g1d(n), 256, 0, st>>>(sc, bufC, bufA, n);
    H = Ho; W = Wo; C = bl.cout;
  }
  run_conv(st, bufA, e.c2w, e.c2b, out, B, H, W, C, 1, 1, 1, outC, 0);
}

extern "C" void kernel_launch(void* const* d_in, const int* in_sizes, int n_in,
                              void* d_out, int out_size, void* d_ws, size_t ws_size,
                              hipStream_t stream)
{
  (void)in_sizes; (void)n_in; (void)out_size; (void)ws_size;

  const float* image1 = (const float*)d_in[0];
  const float* image2 = (const float*)d_in[1];
  EncP cnet, fnet; UpdP up;
  int idx = 2;
  idx = parse_enc(d_in, idx, cnet);   // 'cnet' < 'fnet' < 'update'
  idx = parse_enc(d_in, idx, fnet);
  idx = parse_upd(d_in, idx, up);

  const int B = 2, h = 48, w = 64;
  const size_t NP = (size_t)B * h * w;     // 6144 coarse pixels
  const int P = h * w;                     // 3072 per batch

  // ---- workspace bump allocator (256B aligned) ----
  char* wp = (char*)d_ws;
  auto alloc = [&](size_t nfloat) -> float* {
    float* r = (float*)wp;
    wp += ((nfloat * sizeof(float) + 255) / 256) * 256;
    return r;
  };
  float* f1    = alloc(NP * 256);
  float* f2    = alloc(NP * 256);
  float* cnb   = alloc(NP * 256);
  float* net   = alloc(NP * 128);
  float* inp   = alloc(NP * 128);
  float* pyr0  = alloc(NP * (size_t)P);
  float* pyr1  = alloc(NP * (size_t)(P / 4));
  float* pyr2  = alloc(NP * (size_t)(P / 16));
  float* pyr3  = alloc(NP * (size_t)(P / 64));
  float* bufA  = alloc((size_t)B * 192 * 256 * 64);
  float* bufB  = alloc((size_t)B * 192 * 256 * 64);
  float* bufC  = alloc((size_t)B * 192 * 256 * 64);
  float* corrf = alloc(NP * 324);
  float* cor1  = alloc(NP * 256);
  float* cor2  = alloc(NP * 192);
  float* flo1  = alloc(NP * 128);
  float* flo2  = alloc(NP * 64);
  float* cat0  = alloc(NP * 256);
  float* mo    = alloc(NP * 126);
  float* mo2   = alloc(NP * 128);
  float* xbuf  = alloc(NP * 256);
  float* hx    = alloc(NP * 384);
  float* zc    = alloc(NP * 128);
  float* rc    = alloc(NP * 128);
  float* qc    = alloc(NP * 128);
  float* rh    = alloc(NP * 128);
  float* rhx   = alloc(NP * 384);
  float* dbuf  = alloc(NP * 256);
  float* delta = alloc(NP * 2);
  float* mbuf  = alloc(NP * 256);
  float* mask  = alloc(NP * 576);
  float* flow  = alloc(NP * 2);
  g_kdec       = (int*)alloc(4096);        // k -> (c,kh,kw) LUT scratch

  // ---- feature + context encoders ----
  run_encoder(stream, image1, fnet, 0, f1, 256, bufA, bufB, bufC, B);
  run_encoder(stream, image2, fnet, 0, f2, 256, bufA, bufB, bufC, B);

  // ---- all-pairs correlation + pyramid ----
  {
    dim3 cg(P / TSZ, P / TSZ, B);
    corr_gemm_k<<<cg, 256, 0, stream>>>(f1, f2, pyr0, P, 256, 1.f / 16.f);
  }
  pool_k<<<g1d(NP * (size_t)(P / 4)),  256, 0, stream>>>(pyr0, pyr1, (int)NP, 48, 64);
  pool_k<<<g1d(NP * (size_t)(P / 16)), 256, 0, stream>>>(pyr1, pyr2, (int)NP, 24, 32);
  pool_k<<<g1d(NP * (size_t)(P / 64)), 256, 0, stream>>>(pyr2, pyr3, (int)NP, 12, 16);

  run_encoder(stream, image1, cnet, 1, cnb, 256, bufA, bufB, bufC, B);
  split_cnet_k<<<g1d(NP * 128), 256, 0, stream>>>(cnb, net, inp, NP);
  fill_k<<<g1d(NP * 2), 256, 0, stream>>>(flow, 0.f, NP * 2);

  // ---- 12 update iterations ----
  for (int it = 0; it < 12; ++it) {
    retrieve_k<<<g1d(NP * 81), 256, 0, stream>>>(pyr0, pyr1, pyr2, pyr3,
                                                 flow, corrf, B, h, w);

    run_conv(stream, corrf, up.cc1w, up.cc1b, cor1, B, h, w, 324, 1, 1, 1, 256, 1);
    run_conv(stream, cor1,  up.cc2w, up.cc2b, cor2, B, h, w, 256, 3, 3, 1, 192, 1);
    run_conv(stream, flow,  up.cf1w, up.cf1b, flo1, B, h, w, 2,   7, 7, 1, 128, 1);
    run_conv(stream, flo1,  up.cf2w, up.cf2b, flo2, B, h, w, 128, 3, 3, 1, 64,  1);
    concat_k<<<g1d(NP * 256), 256, 0, stream>>>(cor2, 192, flo2, 64, cat0, NP);
    run_conv(stream, cat0,  up.cmw,  up.cmb,  mo,   B, h, w, 256, 3, 3, 1, 126, 1);
    concat_k<<<g1d(NP * 128), 256, 0, stream>>>(mo, 126, flow, 2, mo2, NP);
    concat_k<<<g1d(NP * 256), 256, 0, stream>>>(inp, 128, mo2, 128, xbuf, NP);

    // separable GRU: pass 0 uses 1x5 convs, pass 1 uses 5x1 convs
    for (int pass = 0; pass < 2; ++pass) {
      const float *zw = pass ? up.z2w : up.z1w, *zb = pass ? up.z2b : up.z1b;
      const float *rw = pass ? up.r2w : up.r1w, *rb = pass ? up.r2b : up.r1b;
      const float *qw = pass ? up.q2w : up.q1w, *qb = pass ? up.q2b : up.q1b;
      int KH = pass ? 5 : 1, KW = pass ? 1 : 5;
      concat_k<<<g1d(NP * 384), 256, 0, stream>>>(net, 128, xbuf, 256, hx, NP);
      run_conv(stream, hx, zw, zb, zc, B, h, w, 384, KH, KW, 1, 128, 0);
      run_conv(stream, hx, rw, rb, rc, B, h, w, 384, KH, KW, 1, 128, 0);
      rh_k<<<g1d(NP * 128), 256, 0, stream>>>(net, rc, rh, NP * 128);
      concat_k<<<g1d(NP * 384), 256, 0, stream>>>(rh, 128, xbuf, 256, rhx, NP);
      run_conv(stream, rhx, qw, qb, qc, B, h, w, 384, KH, KW, 1, 128, 0);
      gru_out_k<<<g1d(NP * 128), 256, 0, stream>>>(net, zc, qc, NP * 128);
    }

    // flow head + mask head
    run_conv(stream, net,  up.fh1w, up.fh1b, dbuf,  B, h, w, 128, 3, 3, 1, 256, 1);
    run_conv(stream, dbuf, up.fh2w, up.fh2b, delta, B, h, w, 256, 3, 3, 1, 2,   0);
    run_conv(stream, net,  up.m1w,  up.m1b,  mbuf,  B, h, w, 128, 3, 3, 1, 256, 1);
    run_conv(stream, mbuf, up.m2w,  up.m2b,  mask,  B, h, w, 256, 1, 1, 1, 576, 0);

    addflow_k<<<g1d(NP * 2), 256, 0, stream>>>(flow, delta, NP * 2);

    float* out_it = (float*)d_out + (size_t)it * B * 384 * 512 * 2;
    upsample_k<<<g1d(NP * 64), 256, 0, stream>>>(flow, mask, out_it, B, h, w);
  }
}
